// Phi4MMAttention_56650618635062
// MI455X (gfx1250) — compile-verified, matrix operand order checked by reading, not run on hardware
//
#include <hip/hip_runtime.h>
#include <hip/hip_bf16.h>

typedef __bf16 bf16;
typedef __attribute__((ext_vector_type(16))) __bf16 v16bf;
typedef __attribute__((ext_vector_type(8)))  float  v8f;
typedef __attribute__((ext_vector_type(4)))  unsigned int v4u;

union FragU { v16bf f; v4u q[2]; };

#define HIDDEN     3072
#define NH         24
#define NKV        8
#define HD         128
#define SEQ        2048
#define NB         2
#define NTOK       (NB*SEQ)      // 4096
#define OPSZ       5120          // NH*HD + 2*NKV*HD
#define LORA_R     256
#define LORA_SC    2.0f          // 512/256
#define ATT_SCALE  0.08838834764831845f  // 128^-0.5

static __device__ __forceinline__ float rmax16(float x) {
#pragma unroll
  for (int m = 1; m < 16; m <<= 1) x = fmaxf(x, __shfl_xor(x, m, 32));
  return x;
}
static __device__ __forceinline__ float rsum16(float x) {
#pragma unroll
  for (int m = 1; m < 16; m <<= 1) x += __shfl_xor(x, m, 32);
  return x;
}

// ---------------------------------------------------------------------------
// CDNA5 async memory->LDS copy (ASYNCcnt-tracked). LDS operand is the low 32
// bits of the flat pointer (workgroup-relative byte offset); global address is
// a 64-bit VGPR pair (GV addressing, no SADDR). Async ops complete in order
// per wave, so s_wait_asynccnt N leaves only the newest N in flight.
// ---------------------------------------------------------------------------
static __device__ __forceinline__ void async_load_b128_raw(unsigned long long ga, unsigned lo) {
  asm volatile("global_load_async_to_lds_b128 %0, %1, off"
               :: "v"(lo), "v"(ga) : "memory");
}
static __device__ __forceinline__ void async_load_b128(const void* gptr, void* lptr) {
  async_load_b128_raw((unsigned long long)(uintptr_t)gptr, (unsigned)(uintptr_t)lptr);
}
static __device__ __forceinline__ void wait_async0() {
  asm volatile("s_wait_asynccnt 0x0" ::: "memory");
}
static __device__ __forceinline__ void wait_async4() {
  asm volatile("s_wait_asynccnt 0x4" ::: "memory");
}

// ---------------------------------------------------------------------------
// fp32 -> bf16 conversion, 8 elems/thread, b128 in / b128 out
// ---------------------------------------------------------------------------
__global__ __launch_bounds__(256) void cvt_f32_to_bf16(
    const float* __restrict__ src, bf16* __restrict__ dst, long n8) {
  long i = (long)blockIdx.x * 256 + threadIdx.x;
  if (i >= n8) return;
  const float4* s4 = (const float4*)src;
  float4 a = s4[2 * i], b = s4[2 * i + 1];
  union { bf16 h[8]; v4u q; } u;
  u.h[0] = (bf16)a.x; u.h[1] = (bf16)a.y; u.h[2] = (bf16)a.z; u.h[3] = (bf16)a.w;
  u.h[4] = (bf16)b.x; u.h[5] = (bf16)b.y; u.h[6] = (bf16)b.z; u.h[7] = (bf16)b.w;
  *(v4u*)(dst + 8 * i) = u.q;
}

// ---------------------------------------------------------------------------
// C[M,N] = alpha * A[M,K] @ B[N,K]^T + beta * Cin     (bf16 in, f32 out)
// 128x128 block tile, 8 waves x (32x64), WMMA bf16.
// 3-stage async-to-LDS pipeline: stage kt+2 each iteration, keep one full
// stage (4 async instrs/wave) outstanding via s_wait_asynccnt 4 so HBM
// latency is hidden behind a full tile of WMMA work. Running 64-bit global
// addresses (+64B/tile) avoid per-iteration address rebuilds.
// Requires M%128==0, N%128==0, K%32==0, K>=64 (true at all call sites).
// ---------------------------------------------------------------------------
__global__ __launch_bounds__(256) void gemm_bf16_nt(
    const bf16* __restrict__ A, const bf16* __restrict__ B,
    const float* __restrict__ Cin, float* __restrict__ Cout,
    int M, int N, int K, float alpha, float beta) {
  __shared__ __align__(16) bf16 As[3][128][40];   // stride 40 bf16 = 80B: 16B aligned, conflict-free
  __shared__ __align__(16) bf16 Bs[3][128][40];   // 3 buffers x 10240B each array
  const int tid  = threadIdx.x;
  const int lane = tid & 31, wave = tid >> 5;
  const int hf = lane >> 4, lr = lane & 15;
  const int m0 = blockIdx.y * 128, n0 = blockIdx.x * 128;
  const int wm = (wave >> 1) * 32, wn = (wave & 1) * 64;
  (void)M;

  v8f acc[2][4];
#pragma unroll
  for (int i = 0; i < 2; i++)
#pragma unroll
    for (int j = 0; j < 4; j++)
#pragma unroll
      for (int e = 0; e < 8; e++) acc[i][j][e] = 0.0f;

  const int KT = K >> 5;

  // per-thread staging slots: running global addresses + buf-0 LDS offsets
  unsigned long long aA[2], aB[2];
  unsigned lA0[2], lB0[2];
#pragma unroll
  for (int s = 0; s < 2; s++) {
    int slot = tid + s * 256, r = slot >> 2, c8 = (slot & 3) << 3;
    aA[s]  = (unsigned long long)(uintptr_t)&A[(size_t)(m0 + r) * K + c8];
    aB[s]  = (unsigned long long)(uintptr_t)&B[(size_t)(n0 + r) * K + c8];
    lA0[s] = (unsigned)(uintptr_t)&As[0][r][c8];
    lB0[s] = (unsigned)(uintptr_t)&Bs[0][r][c8];
  }
  auto stage = [&](int buf) {           // async copy one 128x32 A tile + B tile
    const unsigned boff = (unsigned)buf * (unsigned)(128 * 40 * sizeof(bf16));
#pragma unroll
    for (int s = 0; s < 2; s++) {
      async_load_b128_raw(aA[s], lA0[s] + boff);
      async_load_b128_raw(aB[s], lB0[s] + boff);
      aA[s] += 64;                      // advance one 32-col k-tile
      aB[s] += 64;
    }
  };

  stage(0);                             // tiles 0,1 in flight
  if (KT > 1) stage(1);
  wait_async4();                        // tile 0 landed (tile 1 may be in flight)
  __syncthreads();

  int bufc = 0, bufs = 2;               // compute buffer, next stage buffer
  for (int kt = 0; kt < KT; ++kt) {
    FragU af[2], bfq[4];
#pragma unroll
    for (int i = 0; i < 2; i++) {       // A frag: lane lr = row M; K halves by hf
      int r = wm + i * 16 + lr;
      af[i].q[0] = *(const v4u*)&As[bufc][r][hf * 8];
      af[i].q[1] = *(const v4u*)&As[bufc][r][16 + hf * 8];
    }
#pragma unroll
    for (int j = 0; j < 4; j++) {       // B frag: lane lr = row N (A·B^T symmetry)
      int r = wn + j * 16 + lr;
      bfq[j].q[0] = *(const v4u*)&Bs[bufc][r][hf * 8];
      bfq[j].q[1] = *(const v4u*)&Bs[bufc][r][16 + hf * 8];
    }
#pragma unroll
    for (int i = 0; i < 2; i++)
#pragma unroll
      for (int j = 0; j < 4; j++)
        acc[i][j] = __builtin_amdgcn_wmma_f32_16x16x32_bf16(
            false, af[i].f, false, bfq[j].f, (short)0, acc[i][j], false, false);

    const bool stage_more = (kt + 2) < KT;
    if (stage_more) {                   // overlaps the WMMAs above & next iter
      stage(bufs);
      bufs = (bufs == 2) ? 0 : bufs + 1;
    }
    if ((kt + 1) < KT) {
      if (stage_more) wait_async4();    // tile kt+1 landed, kt+2 still in flight
      else            wait_async0();    // drain tail
      __syncthreads();                  // publish LDS to all waves
    }
    bufc = (bufc == 2) ? 0 : bufc + 1;
  }

  // epilogue: C layout m = e + 8*hf, n = lr within each 16x16 tile.
  // beta test hoisted to a single uniform branch.
  if (beta != 0.0f) {
#pragma unroll
    for (int i = 0; i < 2; i++)
#pragma unroll
      for (int j = 0; j < 4; j++)
#pragma unroll
        for (int e = 0; e < 8; e++) {
          int gr = m0 + wm + i * 16 + hf * 8 + e;
          int gc = n0 + wn + j * 16 + lr;
          size_t o = (size_t)gr * N + gc;
          Cout[o] = alpha * acc[i][j][e] + beta * Cin[o];
        }
  } else {
#pragma unroll
    for (int i = 0; i < 2; i++)
#pragma unroll
      for (int j = 0; j < 4; j++)
#pragma unroll
        for (int e = 0; e < 8; e++) {
          int gr = m0 + wm + i * 16 + hf * 8 + e;
          int gc = n0 + wn + j * 16 + lr;
          Cout[(size_t)gr * N + gc] = alpha * acc[i][j][e];
        }
  }
}

// ---------------------------------------------------------------------------
// RoPE on fp32 qkv rows; splits into head-major bf16 Q[B,NH,S,D],
// K[B,NKV,S,D], V[B,NKV,S,D].
// ---------------------------------------------------------------------------
__global__ __launch_bounds__(256) void rope_split_kernel(
    const float* __restrict__ qkv, const int* __restrict__ pos_ids,
    bf16* __restrict__ q, bf16* __restrict__ k, bf16* __restrict__ v) {
  const int t = blockIdx.x;          // token 0..4095
  const int b = t >> 11, s = t & 2047;
  const float pos = (float)pos_ids[t];
  const float* row = qkv + (size_t)t * OPSZ;
  for (int idx = threadIdx.x; idx < OPSZ; idx += 256) {
    const int head = idx >> 7, d = idx & 127;
    float val = row[idx], outv;
    if (head < NH + NKV && d < 96) {              // rope q heads 0..23, k heads 24..31
      const int j = (d < 48) ? d : d - 48;
      const float ang = pos * __expf(-(float)(2 * j) * (9.210340371976184f / 96.0f));
      const float c = __cosf(ang), sn = __sinf(ang);
      const float other = (d < 48) ? row[idx + 48] : row[idx - 48];
      outv = (d < 48) ? (val * c - other * sn) : (val * c + other * sn);
    } else {
      outv = val;
    }
    const bf16 o = (bf16)outv;
    if (head < NH)
      q[(((size_t)(b * NH + head)) * SEQ + s) * HD + d] = o;
    else if (head < NH + NKV)
      k[(((size_t)(b * NKV + (head - NH))) * SEQ + s) * HD + d] = o;
    else
      v[(((size_t)(b * NKV + (head - NH - NKV))) * SEQ + s) * HD + d] = o;
  }
}

// ---------------------------------------------------------------------------
// Causal flash attention (GQA). Block = (qb,h,b): 128 q rows, 8 waves x 16 rows.
// Q frags register-resident; K staged via async-to-LDS (row-major); V staged
// transposed through registers; online softmax; P routed through a per-wave
// LDS slab into A-fragment layout. Next KV tile prefetched.
// ---------------------------------------------------------------------------
__global__ __launch_bounds__(256) void flash_attn_kernel(
    const bf16* __restrict__ q, const bf16* __restrict__ k,
    const bf16* __restrict__ v, bf16* __restrict__ out) {
  __shared__ __align__(16) bf16 Ks[32][136];   // [kv][d], stride 272B conflict-free
  __shared__ __align__(16) bf16 Vt[128][40];   // [d][kv] transposed
  __shared__ __align__(16) bf16 Ps[8][16][40]; // per-wave P slab
  const int tid  = threadIdx.x;
  const int lane = tid & 31, wave = tid >> 5;
  const int hf = lane >> 4, lr = lane & 15;
  const int qb = blockIdx.x, h = blockIdx.y, b = blockIdx.z;
  const int kvh = h / (NH / NKV);
  const int q0 = qb * 128;
  const bf16* qp = q + ((size_t)((b * NH + h) * SEQ + q0 + wave * 16)) * HD;
  const bf16* kp = k + ((size_t)((b * NKV + kvh) * SEQ)) * HD;
  const bf16* vp = v + ((size_t)((b * NKV + kvh) * SEQ)) * HD;

  FragU qf[4];                                  // Q 16x128 resident: 4 k-steps
#pragma unroll
  for (int kk = 0; kk < 4; kk++) {
    qf[kk].q[0] = *(const v4u*)&qp[lr * HD + kk * 32 + hf * 8];
    qf[kk].q[1] = *(const v4u*)&qp[lr * HD + kk * 32 + 16 + hf * 8];
  }

  float mrow[8], lrow[8];
  v8f acc[8];
#pragma unroll
  for (int e = 0; e < 8; e++) { mrow[e] = -1e30f; lrow[e] = 0.0f; }
#pragma unroll
  for (int t = 0; t < 8; t++)
#pragma unroll
    for (int e = 0; e < 8; e++) acc[t][e] = 0.0f;

  const int kv_end = q0 + 128;                  // causal upper bound (block-uniform)
  const int qrb = q0 + wave * 16 + hf * 8;

  for (int kv0 = 0; kv0 < kv_end; kv0 += 32) {
#pragma unroll
    for (int s = 0; s < 2; s++) {               // K block: async DMA into LDS
      int slot = tid + s * 256, r = slot >> 4, c8 = (slot & 15) << 3;
      async_load_b128(&kp[(size_t)(kv0 + r) * HD + c8], &Ks[r][c8]);
    }
#pragma unroll
    for (int s = 0; s < 2; s++) {               // V block transposed via registers
      int slot = tid + s * 256, r = slot >> 4, c8 = (slot & 15) << 3;
      union { bf16 h2[8]; v4u qq; } u;
      u.qq = *(const v4u*)&vp[(size_t)(kv0 + r) * HD + c8];
#pragma unroll
      for (int j = 0; j < 8; j++) Vt[c8 + j][r] = u.h2[j];
    }
    if (kv0 + 32 < kv_end) {                    // prefetch next KV tile into cache
      const size_t po = (size_t)(kv0 + 32 + (tid >> 4)) * HD + ((tid & 15) << 3);
      __builtin_prefetch(&kp[po], 0, 3);
      __builtin_prefetch(&vp[po], 0, 3);
    }
    wait_async0();
    __syncthreads();

    v8f s0, s1;                                 // scores: two 16x16 tiles (32 kv cols)
#pragma unroll
    for (int e = 0; e < 8; e++) { s0[e] = 0.f; s1[e] = 0.f; }
#pragma unroll
    for (int kk = 0; kk < 4; kk++) {
      FragU b0, b1;
      b0.q[0] = *(const v4u*)&Ks[lr][kk * 32 + hf * 8];
      b0.q[1] = *(const v4u*)&Ks[lr][kk * 32 + 16 + hf * 8];
      b1.q[0] = *(const v4u*)&Ks[16 + lr][kk * 32 + hf * 8];
      b1.q[1] = *(const v4u*)&Ks[16 + lr][kk * 32 + 16 + hf * 8];
      s0 = __builtin_amdgcn_wmma_f32_16x16x32_bf16(false, qf[kk].f, false, b0.f, (short)0, s0, false, false);
      s1 = __builtin_amdgcn_wmma_f32_16x16x32_bf16(false, qf[kk].f, false, b1.f, (short)0, s1, false, false);
    }

    const int c0 = kv0 + lr, c1 = kv0 + 16 + lr;
    float p0[8], p1[8];
#pragma unroll
    for (int e = 0; e < 8; e++) {
      const int qr = qrb + e;
      float a0 = (c0 <= qr) ? s0[e] * ATT_SCALE : -1e30f;
      float a1 = (c1 <= qr) ? s1[e] * ATT_SCALE : -1e30f;
      float nm = fmaxf(mrow[e], rmax16(fmaxf(a0, a1)));
      float sc = __expf(mrow[e] - nm);
      mrow[e] = nm;
      float e0 = __expf(a0 - nm), e1 = __expf(a1 - nm);
      lrow[e] = lrow[e] * sc + rsum16(e0 + e1);
      p0[e] = e0; p1[e] = e1;
#pragma unroll
      for (int t = 0; t < 8; t++) acc[t][e] *= sc;
    }

#pragma unroll
    for (int e = 0; e < 8; e++) {               // P -> per-wave LDS (A-fragment staging)
      Ps[wave][hf * 8 + e][lr]      = (bf16)p0[e];
      Ps[wave][hf * 8 + e][16 + lr] = (bf16)p1[e];
    }
    FragU pf;                                   // same-wave DScnt ordering; no barrier needed
    pf.q[0] = *(const v4u*)&Ps[wave][lr][hf * 8];
    pf.q[1] = *(const v4u*)&Ps[wave][lr][16 + hf * 8];
#pragma unroll
    for (int t = 0; t < 8; t++) {               // acc[16x128] += P(16x32) @ V(32x128)
      FragU vf;
      vf.q[0] = *(const v4u*)&Vt[t * 16 + lr][hf * 8];
      vf.q[1] = *(const v4u*)&Vt[t * 16 + lr][16 + hf * 8];
      acc[t] = __builtin_amdgcn_wmma_f32_16x16x32_bf16(false, pf.f, false, vf.f, (short)0, acc[t], false, false);
    }
    __syncthreads();
  }

#pragma unroll
  for (int e = 0; e < 8; e++) {
    const float rn = 1.0f / lrow[e];
    const size_t row = (size_t)(b * SEQ + q0 + wave * 16 + hf * 8 + e);
#pragma unroll
    for (int t = 0; t < 8; t++)
      out[row * HIDDEN + h * HD + t * 16 + lr] = (bf16)(acc[t][e] * rn);
  }
}

// ---------------------------------------------------------------------------
extern "C" void kernel_launch(void* const* d_in, const int* in_sizes, int n_in,
                              void* d_out, int out_size, void* d_ws, size_t ws_size,
                              hipStream_t stream) {
  (void)in_sizes; (void)n_in; (void)out_size; (void)ws_size;
  const float* hs   = (const float*)d_in[0];
  const float* Wqkv = (const float*)d_in[1];
  const float* Aqkv = (const float*)d_in[2];
  const float* Bqkv = (const float*)d_in[3];
  const float* Wo   = (const float*)d_in[4];
  const float* Ao   = (const float*)d_in[5];
  const float* Bo   = (const float*)d_in[6];
  const int*   pos  = (const int*)d_in[7];
  float* outp = (float*)d_out;

  char* ws = (char*)d_ws;
  size_t off = 0;
  auto alloc = [&](size_t bytes) -> char* {
    char* p = ws + off;
    off += (bytes + 255) & ~(size_t)255;
    return p;
  };
  bf16*  xb    = (bf16*) alloc((size_t)NTOK  * HIDDEN * 2);
  bf16*  wqkvb = (bf16*) alloc((size_t)OPSZ  * HIDDEN * 2);
  bf16*  aqkvb = (bf16*) alloc((size_t)LORA_R* HIDDEN * 2);
  bf16*  bqkvb = (bf16*) alloc((size_t)OPSZ  * LORA_R * 2);
  bf16*  wob   = (bf16*) alloc((size_t)HIDDEN* HIDDEN * 2);
  bf16*  aob   = (bf16*) alloc((size_t)LORA_R* HIDDEN * 2);
  bf16*  bob   = (bf16*) alloc((size_t)HIDDEN* LORA_R * 2);
  float* t1f   = (float*)alloc((size_t)NTOK  * LORA_R * 4);
  bf16*  t1b   = (bf16*) alloc((size_t)NTOK  * LORA_R * 2);
  float* qkvf  = (float*)alloc((size_t)NTOK  * OPSZ   * 4);
  bf16*  qh    = (bf16*) alloc((size_t)NB * NH  * SEQ * HD * 2);
  bf16*  kh    = (bf16*) alloc((size_t)NB * NKV * SEQ * HD * 2);
  bf16*  vh    = (bf16*) alloc((size_t)NB * NKV * SEQ * HD * 2);
  bf16*  attnb = (bf16*) alloc((size_t)NTOK  * HIDDEN * 2);
  float* t2f   = (float*)alloc((size_t)NTOK  * LORA_R * 4);
  bf16*  t2b   = (bf16*) alloc((size_t)NTOK  * LORA_R * 2);

  auto cvt = [&](const float* s, bf16* d, long n) {
    long n8 = n / 8;
    cvt_f32_to_bf16<<<(unsigned)((n8 + 255) / 256), 256, 0, stream>>>(s, d, n8);
  };

  // precision staging
  cvt(hs,   xb,    (long)NTOK   * HIDDEN);
  cvt(Wqkv, wqkvb, (long)OPSZ   * HIDDEN);
  cvt(Aqkv, aqkvb, (long)LORA_R * HIDDEN);
  cvt(Bqkv, bqkvb, (long)OPSZ   * LORA_R);
  cvt(Wo,   wob,   (long)HIDDEN * HIDDEN);
  cvt(Ao,   aob,   (long)LORA_R * HIDDEN);
  cvt(Bo,   bob,   (long)HIDDEN * LORA_R);

  // qkv = x @ Wqkv^T + LORA_SC * (x @ Aqkv^T) @ Bqkv^T
  gemm_bf16_nt<<<dim3(LORA_R / 128, NTOK / 128), 256, 0, stream>>>(
      xb, aqkvb, t1f, t1f, NTOK, LORA_R, HIDDEN, 1.0f, 0.0f);
  cvt(t1f, t1b, (long)NTOK * LORA_R);
  gemm_bf16_nt<<<dim3(OPSZ / 128, NTOK / 128), 256, 0, stream>>>(
      xb, wqkvb, qkvf, qkvf, NTOK, OPSZ, HIDDEN, 1.0f, 0.0f);
  gemm_bf16_nt<<<dim3(OPSZ / 128, NTOK / 128), 256, 0, stream>>>(
      t1b, bqkvb, qkvf, qkvf, NTOK, OPSZ, LORA_R, LORA_SC, 1.0f);

  // RoPE + head split
  rope_split_kernel<<<NTOK, 256, 0, stream>>>(qkvf, pos, qh, kh, vh);

  // causal GQA flash attention -> bf16 [NTOK, HIDDEN]
  flash_attn_kernel<<<dim3(SEQ / 128, NH, NB), 256, 0, stream>>>(qh, kh, vh, attnb);

  // out = attn @ Wo^T + LORA_SC * (attn @ Ao^T) @ Bo^T
  gemm_bf16_nt<<<dim3(LORA_R / 128, NTOK / 128), 256, 0, stream>>>(
      attnb, aob, t2f, t2f, NTOK, LORA_R, HIDDEN, 1.0f, 0.0f);
  cvt(t2f, t2b, (long)NTOK * LORA_R);
  gemm_bf16_nt<<<dim3(HIDDEN / 128, NTOK / 128), 256, 0, stream>>>(
      attnb, wob, outp, outp, NTOK, HIDDEN, HIDDEN, 1.0f, 0.0f);
  gemm_bf16_nt<<<dim3(HIDDEN / 128, NTOK / 128), 256, 0, stream>>>(
      t2b, bob, outp, outp, NTOK, HIDDEN, LORA_R, LORA_SC, 1.0f);
}
